// LocalAttention_38439957300002
// MI455X (gfx1250) — matrix operand order repaired and checked by reading, not compiled, not run on hardware
//
#include <hip/hip_runtime.h>
#include <hip/hip_bf16.h>

// Local causal windowed attention for shape (4,8,4096,64), window=128, look_back=1.
// Per (batch*head, window) block: stage K and V^T in LDS as bf16, then each wave
// computes S^T = K*Q^T and O^T = V^T*P^T with v_wmma_f32_16x16x32_bf16 using
// flash-style online softmax in base-2. Query index lives on lanes in the S^T
// D-layout, so reductions are in-lane + one xor-16 shuffle and the P^T operand
// repacks entirely in registers. Only the last step per wave needs masking.

#define T_SEQ 4096
#define E_DIM 64
#define WS    128
#define JW    256   // keys per window (look_back=1)

typedef __attribute__((ext_vector_type(16))) __bf16 bf16x16;
typedef __attribute__((ext_vector_type(8)))  float  f32x8;

union BReg   { bf16x16 v; unsigned int u[8]; };
union AccReg { f32x8 v; float f[8]; };

__device__ __forceinline__ unsigned int pack2(float lo, float hi) {
#if __has_builtin(__builtin_amdgcn_cvt_pk_bf16_f32)
  auto p = __builtin_amdgcn_cvt_pk_bf16_f32(lo, hi);   // v_cvt_pk_bf16_f32
  return __builtin_bit_cast(unsigned int, p);
#else
  unsigned int ul = __float_as_uint(lo), uh = __float_as_uint(hi);
  ul = (ul + 0x7FFFu + ((ul >> 16) & 1u)) >> 16;       // RNE bf16
  uh = (uh + 0x7FFFu + ((uh >> 16) & 1u)) >> 16;
  return ul | (uh << 16);
#endif
}

__device__ __forceinline__ float fast_exp2(float x) {
#if __has_builtin(__builtin_amdgcn_exp2f)
  return __builtin_amdgcn_exp2f(x);                    // v_exp_f32
#else
  return exp2f(x);
#endif
}

// scale = E^-0.5 * log2(e), softmax carried in base-2 domain
#define SCALE_LOG2E 0.1803368801111204f
#define NEG_BIG     -3.0e38f

__global__ __launch_bounds__(256, 2)
void local_attn_fwd(const float* __restrict__ q, const float* __restrict__ k,
                    const float* __restrict__ v, float* __restrict__ out) {
  __shared__ unsigned short sK [JW * E_DIM];   // bf16 K window  [j][e]   32 KB
  __shared__ unsigned short sVT[E_DIM * JW];   // bf16 V^T       [n][j]   32 KB

  const int tid   = threadIdx.x;
  const int w     = blockIdx.x;                 // window index
  const int bh    = blockIdx.y;                 // batch*head
  const size_t base  = (size_t)bh * T_SEQ * E_DIM;
  const int    grow0 = (w - 1) * WS;            // global row of window key j=0

  // ---- cooperative K load -> LDS bf16 (coalesced float4, zero-fill pad) ----
  #pragma unroll
  for (int it = 0; it < 16; ++it) {
    int idx4 = tid + it * 256;                  // 4096 float4 tasks
    int j  = idx4 >> 4;
    int e4 = idx4 & 15;
    int g  = grow0 + j;
    float4 f = make_float4(0.f, 0.f, 0.f, 0.f);
    if (g >= 0) f = *(const float4*)(k + base + (size_t)g * E_DIM + e4 * 4);
    unsigned int* dst = (unsigned int*)&sK[j * E_DIM + e4 * 4];
    dst[0] = pack2(f.x, f.y);
    dst[1] = pack2(f.z, f.w);
  }
  // ---- cooperative V load -> LDS bf16, transposed [n][j], j-pairs packed ----
  #pragma unroll
  for (int it = 0; it < 8; ++it) {
    int task = tid + it * 256;                  // 2048 tasks: (jp, e4)
    int jp = task >> 4;
    int e4 = task & 15;
    int g0 = grow0 + 2 * jp;                    // even -> g0<0 iff g0+1<0
    float4 f0 = make_float4(0.f,0.f,0.f,0.f), f1 = f0;
    if (g0 >= 0) {
      f0 = *(const float4*)(v + base + (size_t)g0       * E_DIM + e4 * 4);
      f1 = *(const float4*)(v + base + (size_t)(g0 + 1) * E_DIM + e4 * 4);
    }
    float a0[4] = {f0.x, f0.y, f0.z, f0.w};
    float a1[4] = {f1.x, f1.y, f1.z, f1.w};
    #pragma unroll
    for (int qq = 0; qq < 4; ++qq) {
      int n = e4 * 4 + qq;
      *(unsigned int*)&sVT[n * JW + 2 * jp] = pack2(a0[qq], a1[qq]);
    }
  }
  __syncthreads();

  // ---- per-wave 16-query strip ----
  const int m    = __builtin_amdgcn_readfirstlane(tid >> 5); // wave id, uniform
  const int lane = tid & 31;
  const int li   = lane & 15;                  // query column within tile
  const int h    = lane >> 4;                  // lane half
  const int iq   = m * 16 + li;                // local query index in window

  // Q as the B operand of S^T = K*Q^T : e-pair for u[v] is (v<4 ? 2v : 2v+8)+8h
  BReg bq0, bq1;
  const float* qrow = q + base + (size_t)(w * WS + iq) * E_DIM;
  #pragma unroll
  for (int t = 0; t < 4; ++t) {
    float4 f0 = *(const float4*)(qrow + 8 * h + 4 * t + (t >= 2 ? 8 : 0));
    float4 f1 = *(const float4*)(qrow + 32 + 8 * h + 4 * t + (t >= 2 ? 8 : 0));
    bq0.u[2*t]   = pack2(f0.x, f0.y);
    bq0.u[2*t+1] = pack2(f0.z, f0.w);
    bq1.u[2*t]   = pack2(f1.x, f1.y);
    bq1.u[2*t+1] = pack2(f1.z, f1.w);
  }

  AccReg acc[4];                                // O^T tiles: (n = nt*16+vv+8h, i = li)
  #pragma unroll
  for (int nt = 0; nt < 4; ++nt)
    #pragma unroll
    for (int vv = 0; vv < 8; ++vv) acc[nt].f[vv] = 0.f;

  float mrun = NEG_BIG, lrun = 0.f;
  const f32x8 zero8 = {0.f,0.f,0.f,0.f,0.f,0.f,0.f,0.f};

  // One 32-key step; `masked` is a compile-time-foldable constant at call sites.
  auto attn_step = [&](int j0, bool masked) {
    BReg ka;
    AccReg s0, s1;
    #pragma unroll
    for (int vv = 0; vv < 8; ++vv) {            // A e-pairs: (v<4 ? 2v : 2v+8)+8h
      int kb = ((vv < 4) ? 2 * vv : 2 * vv + 8) + 8 * h;
      ka.u[vv] = *(const unsigned int*)&sK[(j0 + li) * E_DIM + kb];
    }
    s0.v = __builtin_amdgcn_wmma_f32_16x16x32_bf16(false, ka.v, false, bq0.v, (short)0, zero8, false, false);
    #pragma unroll
    for (int vv = 0; vv < 8; ++vv) {
      int kb = ((vv < 4) ? 2 * vv : 2 * vv + 8) + 8 * h + 32;
      ka.u[vv] = *(const unsigned int*)&sK[(j0 + li) * E_DIM + kb];
    }
    s0.v = __builtin_amdgcn_wmma_f32_16x16x32_bf16(false, ka.v, false, bq1.v, (short)0, s0.v, false, false);
    #pragma unroll
    for (int vv = 0; vv < 8; ++vv) {
      int kb = ((vv < 4) ? 2 * vv : 2 * vv + 8) + 8 * h;
      ka.u[vv] = *(const unsigned int*)&sK[(j0 + 16 + li) * E_DIM + kb];
    }
    s1.v = __builtin_amdgcn_wmma_f32_16x16x32_bf16(false, ka.v, false, bq0.v, (short)0, zero8, false, false);
    #pragma unroll
    for (int vv = 0; vv < 8; ++vv) {
      int kb = ((vv < 4) ? 2 * vv : 2 * vv + 8) + 8 * h + 32;
      ka.u[vv] = *(const unsigned int*)&sK[(j0 + 16 + li) * E_DIM + kb];
    }
    s1.v = __builtin_amdgcn_wmma_f32_16x16x32_bf16(false, ka.v, false, bq1.v, (short)0, s1.v, false, false);

    // scale (base-2) + optional causal mask + online softmax
    float mloc = NEG_BIG;
    #pragma unroll
    for (int vv = 0; vv < 8; ++vv) {
      float x = s0.f[vv] * SCALE_LOG2E;
      float y = s1.f[vv] * SCALE_LOG2E;
      if (masked) {
        int j = j0 + vv + 8 * h;                // S^T row = key index
        x = (j      > iq + 128) ? NEG_BIG : x;
        y = (j + 16 > iq + 128) ? NEG_BIG : y;
      }
      s0.f[vv] = x; mloc = fmaxf(mloc, x);
      s1.f[vv] = y; mloc = fmaxf(mloc, y);
    }
    mloc = fmaxf(mloc, __shfl_xor(mloc, 16, 32));
    float mnew = fmaxf(mrun, mloc);
    float r = fast_exp2(mrun - mnew);
    mrun = mnew;
    float lloc = 0.f;
    #pragma unroll
    for (int vv = 0; vv < 8; ++vv) {
      s0.f[vv] = fast_exp2(s0.f[vv] - mnew); lloc += s0.f[vv];
      s1.f[vv] = fast_exp2(s1.f[vv] - mnew); lloc += s1.f[vv];
    }
    lloc += __shfl_xor(lloc, 16, 32);
    lrun = lrun * r + lloc;
    #pragma unroll
    for (int nt = 0; nt < 4; ++nt)
      #pragma unroll
      for (int vv = 0; vv < 8; ++vv) acc[nt].f[vv] *= r;

    // repack P^T into the B operand of O^T = V^T * P^T (in-register transpose)
    unsigned int c0[4], c1[4], d0[4], d1[4];
    #pragma unroll
    for (int a = 0; a < 4; ++a) {
      c0[a] = pack2(s0.f[2*a], s0.f[2*a+1]);
      c1[a] = pack2(s1.f[2*a], s1.f[2*a+1]);
      d0[a] = __shfl_xor(c0[a], 16, 32);
      d1[a] = __shfl_xor(c1[a], 16, 32);
    }
    BReg pb;                                    // B j-pairs: (v<4 ? 2v : 2v+8)+8h
    #pragma unroll
    for (int a = 0; a < 4; ++a) {
      pb.u[a]     = c0[a];
      pb.u[a + 4] = c1[a];
    }

    // O^T accumulate: A = V^T strip
    #pragma unroll
    for (int nt = 0; nt < 4; ++nt) {
      BReg av;
      #pragma unroll
      for (int vv = 0; vv < 8; ++vv) {
        int kb = ((vv < 4) ? 2 * vv : 2 * vv + 8) + 8 * h;
        av.u[vv] = *(const unsigned int*)&sVT[(nt * 16 + li) * JW + j0 + kb];
      }
      acc[nt].v = __builtin_amdgcn_wmma_f32_16x16x32_bf16(false, av.v, false, pb.v, (short)0, acc[nt].v, false, false);
    }
  };

  const int jstart = (w == 0) ? 128 : 0;        // skip fully-padded look-back
  int jmaxv = (144 + 16 * m + 31) & ~31;        // past wave's causal diagonal
  const int jend = jmaxv < 256 ? jmaxv : 256;

  // all steps before the last are provably unmasked (j0+31 <= 128 + 16m)
  for (int j0 = jstart; j0 < jend - 32; j0 += 32)
    attn_step(j0, false);
  attn_step(jend - 32, true);                   // diagonal step: causal mask

  // ---- normalize and store (O^T: n = nt*16+vv+8h on regs, i = li on lanes) ----
  const float inv = 1.0f / lrun;
  float* orow = out + base + (size_t)(w * WS + iq) * E_DIM;
  #pragma unroll
  for (int nt = 0; nt < 4; ++nt)
    #pragma unroll
    for (int vv = 0; vv < 8; ++vv)
      orow[nt * 16 + vv + 8 * h] = acc[nt].f[vv] * inv;
}

extern "C" void kernel_launch(void* const* d_in, const int* in_sizes, int n_in,
                              void* d_out, int out_size, void* d_ws, size_t ws_size,
                              hipStream_t stream) {
  (void)n_in; (void)out_size; (void)d_ws; (void)ws_size;
  const float* q = (const float*)d_in[0];
  const float* k = (const float*)d_in[1];
  const float* v = (const float*)d_in[2];
  float* out = (float*)d_out;
  const int bh = in_sizes[0] / (T_SEQ * E_DIM);   // B*H = 32
  dim3 grid(T_SEQ / WS, bh);                      // (32 windows, 32 heads)
  local_attn_fwd<<<grid, 256, 0, stream>>>(q, k, v, out);
}
